// ELiCiT_50087908606684
// MI455X (gfx1250) — compile-verified, matrix operand order checked by reading, not run on hardware
//
#include <hip/hip_runtime.h>
#include <stdint.h>
#include <math.h>

typedef __attribute__((ext_vector_type(16))) _Float16 v16h;
typedef __attribute__((ext_vector_type(4)))  _Float16 v4h;
typedef __attribute__((ext_vector_type(8)))  float    v8f;

#define NFEAT 64
#define QCAND 16
#define DIM0  50000

// ---------------------------------------------------------------------------
// Kernel 1: straight-through quantization of feats -> f16 table in workspace.
// qk[s,f] = candidates[axis(s), f, argmin_q |feats[s,f]-cand|]  (exact in f16:
// candidate values are (2k+1)/32, <=5 significand bits).
// ---------------------------------------------------------------------------
__global__ void elicit_quantize_kernel(const float* __restrict__ feats,
                                       const float* __restrict__ cand,
                                       _Float16* __restrict__ qk, int SF) {
  int gid = blockIdx.x * 256 + threadIdx.x;
  if (gid >= SF) return;
  int s = gid >> 6;            // feature index f = gid & 63 (F=64)
  int f = gid & 63;
  int axis = (s >= DIM0) ? 1 : 0;
  float x = feats[gid];
  const float* cp = cand + ((axis << 6) + f) * QCAND;
  float best = cp[0];
  float bd = fabsf(x - best);
  #pragma unroll
  for (int q = 1; q < QCAND; ++q) {
    float cv = cp[q];
    float d = fabsf(x - cv);
    if (d < bd) { bd = d; best = cv; }   // strict < keeps first min (argmin)
  }
  qk[gid] = (_Float16)best;
}

// ---------------------------------------------------------------------------
// Kernel 2: build split-f16 B matrices (hi + lo correction) for the WMMA
// contraction. K = 256 (8 chunks of 32). K-slot -> (v,f) mapping chosen so
// that a lane's A slots cover 4 contiguous features:
//   chunk c, local kl: grp = kl>>3, f = 8c + {0,4,2,6}[grp] + ((kl&7)>>2),
//   v = kl&3.
// B[k, n]: col 0 accumulates features f<32 ("a"), col 1 features f>=32 ("b").
// Lane layout mirrors the 16-bit A layout: lane half h holds K {0-7,16-23}
// (h=0) or {8-15,24-31} (h=1); VGPR j holds K pair (2j',2j'+1), even K in
// bits [15:0].
// ---------------------------------------------------------------------------
__global__ void elicit_prep_bmat_kernel(const float* __restrict__ values,
                                        uint32_t* __restrict__ bmat) {
  int t = threadIdx.x;            // 256 threads -> (chunk, lane)
  int c = t >> 5;
  int lane = t & 31;
  int h = lane >> 4;
  int n = lane & 15;
  const int fofs[4] = {0, 4, 2, 6};
  #pragma unroll
  for (int j = 0; j < 8; ++j) {
    uint32_t dwh = 0, dwl = 0;
    #pragma unroll
    for (int e = 0; e < 2; ++e) {
      int kl = (j < 4) ? ((h ? 8 : 0) + 2 * j + e)
                       : ((h ? 24 : 16) + 2 * (j - 4) + e);
      int grp = kl >> 3;
      int within = kl & 7;
      int f = 8 * c + fofs[grp] + (within >> 2);
      int v = kl & 3;
      float wv = 0.0f;
      if ((f < 32 && n == 0) || (f >= 32 && n == 1)) wv = values[v * NFEAT + f];
      _Float16 hi = (_Float16)wv;
      _Float16 lo = (_Float16)(wv - (float)hi);
      dwh |= ((uint32_t)__builtin_bit_cast(unsigned short, hi)) << (16 * e);
      dwl |= ((uint32_t)__builtin_bit_cast(unsigned short, lo)) << (16 * e);
    }
    int base = (c * 32 + lane) * 8 + j;
    bmat[base]        = dwh;   // B_hi
    bmat[2048 + base] = dwl;   // B_lo (f16 correction term)
  }
}

// ---------------------------------------------------------------------------
// Kernel 3: main query kernel. 256 threads = 8 waves; each wave computes 16
// queries via 16 chained v_wmma_f32_16x16x32_f16 (8 K-chunks x {hi,lo}).
// A[m=query, k=(v,f)] = bilinear corner weight, exact in f16.
// D[m,0]=a, D[m,1]=b; lanes 0/16 hold a, lanes 1/17 hold b (shfl_xor to mix).
// ---------------------------------------------------------------------------
__global__ void elicit_main_kernel(const _Float16* __restrict__ qk,
                                   const uint32_t* __restrict__ bmat,
                                   const float* __restrict__ scale,
                                   const float* __restrict__ bias,
                                   const int* __restrict__ idx0,
                                   const int* __restrict__ idx1,
                                   float* __restrict__ out, int nq) {
  __shared__ __align__(32) uint32_t bsh[4096];   // 16 KB: B_hi + B_lo
  for (int i = threadIdx.x; i < 4096; i += 256) bsh[i] = bmat[i];
  __syncthreads();

  int tid  = threadIdx.x;
  int wave = tid >> 5;
  int lane = tid & 31;
  int m    = lane & 15;        // query within the wave's 16-query tile
  int h    = lane >> 4;        // lane half selects K sub-ranges / features
  int qbase = blockIdx.x * 128 + wave * 16;
  int q = qbase + m;
  if (q >= nq) q = nq - 1;     // clamp: keep EXEC all-ones for WMMA

  int i0 = idx0[q];
  int i1 = idx1[q];
  const _Float16* r0 = qk + (size_t)i0 * NFEAT;
  const _Float16* r1 = qk + ((size_t)DIM0 + (size_t)i1) * NFEAT;
  __builtin_prefetch(r0, 0, 0);   // global_prefetch_b8 for the gathered rows
  __builtin_prefetch(r1, 0, 0);

  v8f acc = {0.f, 0.f, 0.f, 0.f, 0.f, 0.f, 0.f, 0.f};
  #pragma unroll
  for (int c = 0; c < 8; ++c) {
    int fb = 8 * c + 4 * h;                      // 4 contiguous features
    v4h a0 = *(const v4h*)(r0 + fb);             // 8B gathers (exact f16)
    v4h a1 = *(const v4h*)(r1 + fb);
    v16h A;
    #pragma unroll
    for (int p = 0; p < 4; ++p) {
      _Float16 u  = a0[p];
      _Float16 w  = a1[p];
      _Float16 iu = (_Float16)1.0f - u;          // exact: (32-odd)/32
      _Float16 iw = (_Float16)1.0f - w;
      A[4 * p + 0] = u * w;                      // corner v=0  (exact in f16)
      A[4 * p + 1] = u * iw;                     // corner v=1
      A[4 * p + 2] = iu * w;                     // corner v=2
      A[4 * p + 3] = iu * iw;                    // corner v=3
    }
    v16h Bh = *(const v16h*)&bsh[(c * 32 + lane) * 8];
    v16h Bl = *(const v16h*)&bsh[2048 + (c * 32 + lane) * 8];
    acc = __builtin_amdgcn_wmma_f32_16x16x32_f16(false, A, false, Bh,
                                                 (short)0, acc, false, false);
    acc = __builtin_amdgcn_wmma_f32_16x16x32_f16(false, A, false, Bl,
                                                 (short)0, acc, false, false);
  }

  float es = expf(scale[0]);
  float bb = bias[0];
  // D layout: VGPR r, lanes 0-15 -> (M=r, N=lane); lanes 16-31 -> (M=8+r).
  // a = column N=0 (lanes 0,16); b = column N=1 (lanes 1,17).
  #pragma unroll
  for (int r = 0; r < 8; ++r) {
    float other = __shfl_xor(acc[r], 1, 32);     // pair a-lane with b-lane
    if ((lane & 15) == 0) {
      int qq = qbase + h * 8 + r;
      if (qq < nq) out[qq] = acc[r] * tanhf(other) * es + bb;
    }
  }
}

// ---------------------------------------------------------------------------
extern "C" void kernel_launch(void* const* d_in, const int* in_sizes, int n_in,
                              void* d_out, int out_size, void* d_ws, size_t ws_size,
                              hipStream_t stream) {
  const float* values = (const float*)d_in[0];   // [1,4,64]
  const float* feats  = (const float*)d_in[1];   // [100000,64]
  const float* cand   = (const float*)d_in[2];   // [2,64,16]
  const float* scale  = (const float*)d_in[3];   // [1]
  const float* bias   = (const float*)d_in[4];   // [1]
  const int*   idx0   = (const int*)d_in[5];     // [N]
  const int*   idx1   = (const int*)d_in[6];     // [N]
  float* out = (float*)d_out;

  int SF = in_sizes[1];                          // 6,400,000
  int nq = in_sizes[5];                          // 1,000,000

  _Float16* qk   = (_Float16*)d_ws;                                  // 12.8 MB
  uint32_t* bmat = (uint32_t*)((char*)d_ws + (size_t)SF * sizeof(_Float16)); // 16 KB

  elicit_quantize_kernel<<<(SF + 255) / 256, 256, 0, stream>>>(feats, cand, qk, SF);
  elicit_prep_bmat_kernel<<<1, 256, 0, stream>>>(values, bmat);
  elicit_main_kernel<<<(nq + 127) / 128, 256, 0, stream>>>(qk, bmat, scale, bias,
                                                           idx0, idx1, out, nq);
}
